// ShapeTransformer_32598801777000
// MI455X (gfx1250) — compile-verified, matrix operand order checked by reading, not run on hardware
//
#include <hip/hip_runtime.h>
#include <hip/hip_bf16.h>
#include <math.h>

typedef __attribute__((ext_vector_type(16))) __bf16 v16bf;
typedef __attribute__((ext_vector_type(8)))  float  v8f;

#define WMMA_BF16(a, b, c) \
  __builtin_amdgcn_wmma_f32_16x16x32_bf16(false, (a), false, (b), (short)0, (c), false, false)

// Starting K index of the 2-element group held in "VGPR j" of a 16-bit A/B
// operand (CDNA5 ISA, 16-bit A-matrix 16x32 layout):
//   VGPR 0..3: lanes0-15 K=0..7, lanes16-31 K=8..15 ; VGPR 4..7: +16 / +24
__device__ __forceinline__ int kgrp(int j, int half) {
  return ((j < 4) ? (2 * j) : (16 + 2 * (j - 4))) + (half << 3);
}

// ---- LDS layout (dynamic) ----
// f32 words:
//   xl[1040], hs[144*16], aob[144*16], wproj[128],
//   bproj[16], bin[48], bout[16], ln1g,ln1b,w1,w2,b2,ln2g,ln2b (7*16), b1(+pad 4), red[256]
// bf16 elems:
//   hb[144*32], winb[48*16], woutb[256], qb/kb/vb/cxb[144*16 each], pb[8*16*160]
#define F32_WORDS  (1040 + 2304 + 2304 + 128 + 16 + 48 + 16 + 112 + 4 + 256)
#define BF16_ELEMS (144 * 32 + 48 * 16 + 256 + 4 * 144 * 16 + 8 * 16 * 160)
#define SHMEM_BYTES (F32_WORDS * 4 + BF16_ELEMS * 2)

__global__ __launch_bounds__(256)
void ShapeTransformer_kernel(const float* __restrict__ x,
                             const float* __restrict__ w_proj,
                             const float* __restrict__ b_proj,
                             const float* __restrict__ w_in,
                             const float* __restrict__ b_in,
                             const float* __restrict__ w_out,
                             const float* __restrict__ b_out,
                             const float* __restrict__ ln1_g,
                             const float* __restrict__ ln1_b,
                             const float* __restrict__ w1,
                             const float* __restrict__ b1,
                             const float* __restrict__ w2,
                             const float* __restrict__ b2,
                             const float* __restrict__ ln2_g,
                             const float* __restrict__ ln2_b,
                             float* __restrict__ out) {
  extern __shared__ char smem[];
  float*  xl      = (float*)smem;            // 1040
  float*  hs      = xl + 1040;               // 144*16 (f32 h, then LN2 output)
  float*  aob     = hs + 2304;               // 144*16 (f32 attn_out incl. bias)
  float*  wproj_l = aob + 2304;              // 16*8
  float*  bproj_l = wproj_l + 128;           // 16
  float*  bin_l   = bproj_l + 16;            // 48
  float*  bout_l  = bin_l + 48;              // 16
  float*  ln1g_l  = bout_l + 16;
  float*  ln1b_l  = ln1g_l + 16;
  float*  w1_l    = ln1b_l + 16;
  float*  w2_l    = w1_l + 16;
  float*  b2_l    = w2_l + 16;
  float*  ln2g_l  = b2_l + 16;
  float*  ln2b_l  = ln2g_l + 16;
  float*  b1_l    = ln2b_l + 16;             // 1 (+3 pad)
  float*  red     = b1_l + 4;                // 16 parts x 16 cols
  __bf16* hb      = (__bf16*)(red + 256);    // 144*32 (K padded to 32, zeroed)
  __bf16* winb    = hb + 144 * 32;           // 48*16
  __bf16* woutb   = winb + 768;              // 16*16
  __bf16* qb      = woutb + 256;             // 144*16
  __bf16* kbuf    = qb + 2304;               // 144*16
  __bf16* vbuf    = kbuf + 2304;             // 144*16
  __bf16* cxb     = vbuf + 2304;             // 144*16 (ctx, bf16)
  __bf16* pb      = cxb + 2304;              // 8 waves * 16 * 160

  const int tid  = threadIdx.x;
  const int lane = tid & 31;
  const int wave = tid >> 5;
  const int half = (lane >> 4) & 1;
  const int m16  = lane & 15;
  const int bidx = blockIdx.x;

  // ---------------- Stage 0: stage x, weights; embed + pos-enc ----------------
  for (int i = tid; i < 1040; i += 256)
    xl[i] = (i < 1039) ? x[(size_t)bidx * 1039 + i] : 0.f;
  for (int i = tid; i < 144 * 32; i += 256) hb[i] = (__bf16)0.f;
  for (int i = tid; i < 768; i += 256) winb[i] = (__bf16)w_in[i];
  woutb[tid] = (__bf16)w_out[tid];  // blockDim == 256
  if (tid < 128) wproj_l[tid] = w_proj[tid];
  if (tid < 48)  bin_l[tid] = b_in[tid];
  if (tid < 16) {
    bproj_l[tid] = b_proj[tid]; bout_l[tid] = b_out[tid];
    ln1g_l[tid] = ln1_g[tid];   ln1b_l[tid] = ln1_b[tid];
    w1_l[tid] = w1[tid];        w2_l[tid] = w2[tid];  b2_l[tid] = b2[tid];
    ln2g_l[tid] = ln2_g[tid];   ln2b_l[tid] = ln2_b[tid];
  }
  if (tid == 0) b1_l[0] = b1[0];
  __syncthreads();

  for (int i = tid; i < 144 * 16; i += 256) {
    int s = i >> 4, e = i & 15;
    float v = 0.f;
    if (s < 130) {
      v = bproj_l[e];
#pragma unroll
      for (int p = 0; p < 8; ++p) v += xl[s * 8 + p] * wproj_l[e * 8 + p];
      // pos encoding: ln(10000)/16 = 0.5756462732485115
      float freq = expf(-(float)(e & ~1) * 0.5756462732485115f);
      float ang = (float)s * freq;
      v += (e & 1) ? cosf(ang) : sinf(ang);
    }
    hs[i] = v;
    hb[s * 32 + e] = (__bf16)v;  // rows >=130 stay 0, cols 16..31 stay 0
  }
  __syncthreads();

  // ---------------- Stage 1: QKV = h @ w_in^T + b_in (WMMA) ----------------
  for (int t = wave; t < 9; t += 8) {
    v16bf a;
#pragma unroll
    for (int j = 0; j < 8; ++j) {
      int k0 = kgrp(j, half);
      a[2 * j]     = hb[(t * 16 + m16) * 32 + k0];
      a[2 * j + 1] = hb[(t * 16 + m16) * 32 + k0 + 1];
    }
#pragma unroll
    for (int o = 0; o < 3; ++o) {
      v16bf bm;
#pragma unroll
      for (int j = 0; j < 8; ++j) {
        int k0 = kgrp(j, half);  // even
        bm[2 * j]     = (k0 < 16) ? winb[(o * 16 + m16) * 16 + k0]     : (__bf16)0.f;
        bm[2 * j + 1] = (k0 < 16) ? winb[(o * 16 + m16) * 16 + k0 + 1] : (__bf16)0.f;
      }
      v8f acc = {0.f, 0.f, 0.f, 0.f, 0.f, 0.f, 0.f, 0.f};
      acc = WMMA_BF16(a, bm, acc);
      __bf16* dst = (o == 0) ? qb : (o == 1) ? kbuf : vbuf;
#pragma unroll
      for (int r = 0; r < 8; ++r) {
        int row = t * 16 + r + 8 * half;
        dst[row * 16 + m16] = (__bf16)(acc[r] + bin_l[o * 16 + m16]);
      }
    }
  }
  __syncthreads();

  // ---------------- Stage 2: attention + out-proj (per-wave, WMMA) ----------------
  __bf16* pw = pb + wave * (16 * 160);
  for (int t = wave; t < 9; t += 8) {
    v16bf aq;
#pragma unroll
    for (int j = 0; j < 8; ++j) {
      int k0 = kgrp(j, half);
      aq[2 * j]     = (k0 < 16) ? qb[(t * 16 + m16) * 16 + k0]     : (__bf16)0.f;
      aq[2 * j + 1] = (k0 < 16) ? qb[(t * 16 + m16) * 16 + k0 + 1] : (__bf16)0.f;
    }
    float sc[9][8];
#pragma unroll
    for (int j2 = 0; j2 < 9; ++j2) {
      v16bf bk;
#pragma unroll
      for (int j = 0; j < 8; ++j) {
        int k0 = kgrp(j, half);
        bk[2 * j]     = (k0 < 16) ? kbuf[(j2 * 16 + m16) * 16 + k0]     : (__bf16)0.f;
        bk[2 * j + 1] = (k0 < 16) ? kbuf[(j2 * 16 + m16) * 16 + k0 + 1] : (__bf16)0.f;
      }
      v8f acc = {0.f, 0.f, 0.f, 0.f, 0.f, 0.f, 0.f, 0.f};
      acc = WMMA_BF16(aq, bk, acc);
      int col = j2 * 16 + m16;
#pragma unroll
      for (int r = 0; r < 8; ++r)
        sc[j2][r] = (col < 130) ? acc[r] * 0.25f : -3.0e38f;
    }
    // row softmax (rows live in 16-lane halves of the D layout)
#pragma unroll
    for (int r = 0; r < 8; ++r) {
      float m = -3.0e38f;
#pragma unroll
      for (int j2 = 0; j2 < 9; ++j2) m = fmaxf(m, sc[j2][r]);
#pragma unroll
      for (int off = 1; off < 16; off <<= 1) m = fmaxf(m, __shfl_xor(m, off, 32));
      float sum = 0.f;
#pragma unroll
      for (int j2 = 0; j2 < 9; ++j2) {
        float p = __expf(sc[j2][r] - m);
        p = (sc[j2][r] <= -1.0e38f) ? 0.f : p;
        sc[j2][r] = p;
        sum += p;
      }
#pragma unroll
      for (int off = 1; off < 16; off <<= 1) sum += __shfl_xor(sum, off, 32);
      float inv = 1.f / sum;
      int row = r + 8 * half;
#pragma unroll
      for (int j2 = 0; j2 < 9; ++j2)
        pw[row * 160 + j2 * 16 + m16] = (__bf16)(sc[j2][r] * inv);
      pw[row * 160 + 144 + m16] = (__bf16)0.f;  // pad cols 144..159
    }
    // ctx = P @ V  (5 accumulating WMMAs, K=32 each)
    v8f cacc = {0.f, 0.f, 0.f, 0.f, 0.f, 0.f, 0.f, 0.f};
#pragma unroll
    for (int c = 0; c < 5; ++c) {
      v16bf ap, bv;
#pragma unroll
      for (int j = 0; j < 8; ++j) {
        int k0 = kgrp(j, half);
        ap[2 * j]     = pw[m16 * 160 + c * 32 + k0];
        ap[2 * j + 1] = pw[m16 * 160 + c * 32 + k0 + 1];
        int kg = c * 32 + k0;
        bv[2 * j]     = (kg < 144)     ? vbuf[kg * 16 + m16]       : (__bf16)0.f;
        bv[2 * j + 1] = (kg + 1 < 144) ? vbuf[(kg + 1) * 16 + m16] : (__bf16)0.f;
      }
      cacc = WMMA_BF16(ap, bv, cacc);
    }
    // round-trip ctx through LDS (wave-local rows) to re-shape D-layout -> A-layout
#pragma unroll
    for (int r = 0; r < 8; ++r)
      cxb[(t * 16 + r + 8 * half) * 16 + m16] = (__bf16)cacc[r];
    // attn_out = ctx @ w_out^T + b_out (WMMA)
    v16bf ac, bw;
#pragma unroll
    for (int j = 0; j < 8; ++j) {
      int k0 = kgrp(j, half);
      ac[2 * j]     = (k0 < 16) ? cxb[(t * 16 + m16) * 16 + k0]     : (__bf16)0.f;
      ac[2 * j + 1] = (k0 < 16) ? cxb[(t * 16 + m16) * 16 + k0 + 1] : (__bf16)0.f;
      bw[2 * j]     = (k0 < 16) ? woutb[m16 * 16 + k0]              : (__bf16)0.f;
      bw[2 * j + 1] = (k0 < 16) ? woutb[m16 * 16 + k0 + 1]          : (__bf16)0.f;
    }
    v8f oacc = {0.f, 0.f, 0.f, 0.f, 0.f, 0.f, 0.f, 0.f};
    oacc = WMMA_BF16(ac, bw, oacc);
#pragma unroll
    for (int r = 0; r < 8; ++r)
      aob[(t * 16 + r + 8 * half) * 16 + m16] = oacc[r] + bout_l[m16];
  }
  __syncthreads();

  // ---------------- Stage 3: residual + LN1 + rank-1 FFN + LN2 (one thread/row) ----------------
  if (tid < 130) {
    int s = tid;
    float y[16];
    float mean = 0.f;
#pragma unroll
    for (int e = 0; e < 16; ++e) {
      y[e] = hs[s * 16 + e] + aob[s * 16 + e];
      mean += y[e];
    }
    mean *= (1.f / 16.f);
    float var = 0.f;
#pragma unroll
    for (int e = 0; e < 16; ++e) { float d = y[e] - mean; var += d * d; }
    var *= (1.f / 16.f);
    float rstd = rsqrtf(var + 1e-5f);
    float h1[16];
    float ffin = b1_l[0];
#pragma unroll
    for (int e = 0; e < 16; ++e) {
      h1[e] = (y[e] - mean) * rstd * ln1g_l[e] + ln1b_l[e];
      ffin += h1[e] * w1_l[e];
    }
    float ff = fmaxf(ffin, 0.f);
    float z[16];
    float mean2 = 0.f;
#pragma unroll
    for (int e = 0; e < 16; ++e) {
      z[e] = h1[e] + ff * w2_l[e] + b2_l[e];
      mean2 += z[e];
    }
    mean2 *= (1.f / 16.f);
    float var2 = 0.f;
#pragma unroll
    for (int e = 0; e < 16; ++e) { float d = z[e] - mean2; var2 += d * d; }
    var2 *= (1.f / 16.f);
    float rstd2 = rsqrtf(var2 + 1e-5f);
#pragma unroll
    for (int e = 0; e < 16; ++e)
      hs[s * 16 + e] = (z[e] - mean2) * rstd2 * ln2g_l[e] + ln2b_l[e];
  }
  __syncthreads();

  // ---------------- Stage 4: column mean over S (parallel) ----------------
  {
    int e = tid & 15, part = tid >> 4;  // 16 parts x 16 cols
    float acc = 0.f;
    for (int s = part; s < 130; s += 16) acc += hs[s * 16 + e];
    red[part * 16 + e] = acc;
  }
  __syncthreads();
  if (tid < 16) {
    float acc = 0.f;
#pragma unroll
    for (int p = 0; p < 16; ++p) acc += red[p * 16 + tid];
    out[(size_t)bidx * 16 + tid] = acc * (1.f / 130.f);
  }
}

extern "C" void kernel_launch(void* const* d_in, const int* in_sizes, int n_in,
                              void* d_out, int out_size, void* d_ws, size_t ws_size,
                              hipStream_t stream) {
  const float* x      = (const float*)d_in[0];
  const float* w_proj = (const float*)d_in[1];
  const float* b_proj = (const float*)d_in[2];
  const float* w_in   = (const float*)d_in[3];
  const float* b_in   = (const float*)d_in[4];
  const float* w_out  = (const float*)d_in[5];
  const float* b_out  = (const float*)d_in[6];
  const float* ln1_g  = (const float*)d_in[7];
  const float* ln1_b  = (const float*)d_in[8];
  const float* w1     = (const float*)d_in[9];
  const float* b1     = (const float*)d_in[10];
  const float* w2     = (const float*)d_in[11];
  const float* b2     = (const float*)d_in[12];
  const float* ln2_g  = (const float*)d_in[13];
  const float* ln2_b  = (const float*)d_in[14];
  float* out = (float*)d_out;

  int B = in_sizes[0] / 1039;

  (void)hipFuncSetAttribute((const void*)ShapeTransformer_kernel,
                            hipFuncAttributeMaxDynamicSharedMemorySize,
                            (int)SHMEM_BYTES);
  ShapeTransformer_kernel<<<B, 256, SHMEM_BYTES, stream>>>(
      x, w_proj, b_proj, w_in, b_in, w_out, b_out,
      ln1_g, ln1_b, w1, b1, w2, b2, ln2_g, ln2_b, out);
}
